// GCN_83099027243500
// MI455X (gfx1250) — matrix operand order repaired and checked by reading, not compiled, not run on hardware
//
#include <hip/hip_runtime.h>
#include <cstdint>

typedef __attribute__((ext_vector_type(16))) _Float16 v16h;
typedef __attribute__((ext_vector_type(8)))  float    v8f;

#define BN_EPS 1e-5f

// ---------------- degree / normalization ----------------
__global__ void k_deg_init(float* __restrict__ deg, int n) {
    int i = blockIdx.x * blockDim.x + threadIdx.x;
    if (i < n) deg[i] = 1.0f;                    // self-loop
}
__global__ void k_deg_accum(const int* __restrict__ dst, float* __restrict__ deg, int e) {
    int i = blockIdx.x * blockDim.x + threadIdx.x;
    if (i < e) atomicAdd(&deg[dst[i]], 1.0f);
}
__global__ void k_deg_rsqrt(float* __restrict__ deg, int n) {
    int i = blockIdx.x * blockDim.x + threadIdx.x;
    if (i < n) deg[i] = rsqrtf(deg[i]);          // deg -> dinv in place
}

// ---------------- pack W[fin,fout] (f32) -> fragment-ordered f16, zero-padded ----------------
// Fragment layout: Wf[(((ks*tiles_n + ct)*32 + lane)*16 + i]
//   lane: n = ct*16 + (lane&15), hi = lane>>4 ;  element i -> K = ks*32 + 8*hi + (i&7) + 16*(i>>3)
__global__ void k_pack_w(const float* __restrict__ W, _Float16* __restrict__ Wf,
                         int fout, int tiles_n, int total)
{
    int idx = blockIdx.x * blockDim.x + threadIdx.x;
    if (idx >= total) return;
    int i    = idx & 15;
    int lane = (idx >> 4) & 31;
    int t    = idx >> 9;
    int ct   = t % tiles_n;
    int ks   = t / tiles_n;
    int hi   = lane >> 4;
    int n    = (ct << 4) + (lane & 15);
    int k    = ks * 32 + 8 * hi + (i & 7) + 16 * (i >> 3);
    Wf[idx]  = (n < fout) ? (_Float16)W[(size_t)k * fout + n] : (_Float16)0.0f;
}

// ---------------- WMMA GEMM:  H[nrows,fout] = X[nrows, KSTEPS*32] @ W ----------------
// One wave = one 16x16 tile, 8 waves/block. Packed W staged in LDS via async-to-LDS DMA.
template<int KSTEPS>
__global__ __launch_bounds__(256) void k_gemm_wmma(
    const float* __restrict__ X, const _Float16* __restrict__ Wf,
    float* __restrict__ H, int nrows, int fout, int tiles_n)
{
    __shared__ __align__(32) _Float16 smem[KSTEPS * 8 * 512];   // up to 32 KB

    // ---- cooperative async copy of packed W into LDS ----
    const int nbytes = KSTEPS * tiles_n * 1024;                 // bytes of Wf
    const unsigned lds_base = (unsigned)(uintptr_t)(&smem[0]);  // low 32 bits = LDS offset
    for (int byte = threadIdx.x * 16; byte < nbytes; byte += 256 * 16) {
        unsigned lds = lds_base + byte;
        const char* g = (const char*)Wf + byte;
        asm volatile("global_load_async_to_lds_b128 %0, %1, off"
                     :: "v"(lds), "v"(g) : "memory");
    }
    asm volatile("s_wait_asynccnt 0x0" ::: "memory");
    __syncthreads();

    const int wave = threadIdx.x >> 5;
    const int lane = threadIdx.x & 31;
    const int tiles_m = (nrows + 15) >> 4;
    const int tile_id = blockIdx.x * 8 + wave;
    if (tile_id >= tiles_m * tiles_n) return;    // wave-uniform; EXEC all-1s at WMMA
    const int rt   = tile_id / tiles_n;
    const int ct   = tile_id - rt * tiles_n;
    const int row0 = rt << 4;

    const int m  = lane & 15;
    const int hi = lane >> 4;
    const int n  = (ct << 4) + m;

    int ar = row0 + m; if (ar >= nrows) ar = nrows - 1;
    const float* Arow = X + (size_t)ar * (KSTEPS * 32);

    v8f acc = {};
    #pragma unroll
    for (int ks = 0; ks < KSTEPS; ++ks) {
        // A fragment (16-bit 16x32 layout): lane<16 K=0..7,16..23 ; lane>=16 K=8..15,24..31
        const float* p = Arow + ks * 32 + 8 * hi;
        float4 a0 = *(const float4*)(p);
        float4 a1 = *(const float4*)(p + 4);
        float4 a2 = *(const float4*)(p + 16);
        float4 a3 = *(const float4*)(p + 20);
        v16h a;
        a[0]=(_Float16)a0.x;  a[1]=(_Float16)a0.y;  a[2]=(_Float16)a0.z;  a[3]=(_Float16)a0.w;
        a[4]=(_Float16)a1.x;  a[5]=(_Float16)a1.y;  a[6]=(_Float16)a1.z;  a[7]=(_Float16)a1.w;
        a[8]=(_Float16)a2.x;  a[9]=(_Float16)a2.y;  a[10]=(_Float16)a2.z; a[11]=(_Float16)a2.w;
        a[12]=(_Float16)a3.x; a[13]=(_Float16)a3.y; a[14]=(_Float16)a3.z; a[15]=(_Float16)a3.w;

        const v16h b = *(const v16h*)&smem[((ks * tiles_n + ct) * 32 + lane) * 16];

        acc = __builtin_amdgcn_wmma_f32_16x16x32_f16(
            false, a, false, b, (short)0, acc, false, false);
    }

    if (n < fout) {
        #pragma unroll
        for (int r = 0; r < 8; ++r) {
            int row = row0 + r + 8 * hi;
            if (row < nrows) H[(size_t)row * fout + n] = acc[r];
        }
    }
}

// ---------------- self-loop init: agg = h * dinv^2 (overwrites target) ----------------
__global__ void k_selfloop_init(const float* __restrict__ h, const float* __restrict__ dinv,
                                float* __restrict__ agg, long long total, int f)
{
    long long idx = (long long)blockIdx.x * blockDim.x + threadIdx.x;
    if (idx < total) {
        int node = (int)(idx / f);
        float di = dinv[node];
        agg[idx] = h[idx] * di * di;
    }
}

// ---------------- edge scatter, f==128: wave/edge, lane owns 4 contiguous features ----------------
__global__ __launch_bounds__(256) void k_edge_scatter128(
    const int* __restrict__ src, const int* __restrict__ dst,
    const float* __restrict__ dinv, const float* __restrict__ h,
    float* __restrict__ agg, int e)
{
    int gid  = blockIdx.x * blockDim.x + threadIdx.x;
    int wid  = gid >> 5;
    int lane = gid & 31;
    if (wid >= e) return;
    int s = src[wid], d = dst[wid];
    float c = dinv[s] * dinv[d];
    const float4 hv = *(const float4*)(h + (size_t)s * 128 + lane * 4);
    float* ad = agg + (size_t)d * 128 + lane * 4;
    atomicAdd(ad + 0, hv.x * c);
    atomicAdd(ad + 1, hv.y * c);
    atomicAdd(ad + 2, hv.z * c);
    atomicAdd(ad + 3, hv.w * c);
}

// ---------------- edge scatter, small f: one thread per edge ----------------
__global__ void k_edge_scatter_small(
    const int* __restrict__ src, const int* __restrict__ dst,
    const float* __restrict__ dinv, const float* __restrict__ h,
    float* __restrict__ agg, int e, int f)
{
    int i = blockIdx.x * blockDim.x + threadIdx.x;
    if (i >= e) return;
    int s = src[i], d = dst[i];
    float c = dinv[s] * dinv[d];
    const float* hs = h + (size_t)s * f;
    float* ad = agg + (size_t)d * f;
    for (int j = 0; j < f; ++j)
        atomicAdd(&ad[j], hs[j] * c);
}

// ---------------- fused bias + BN(eval) + ReLU (in-place safe) ----------------
__global__ void k_bias_bn_relu(const float* __restrict__ agg, const float* __restrict__ b,
                               const float* __restrict__ g, const float* __restrict__ be,
                               const float* __restrict__ m, const float* __restrict__ v,
                               float* __restrict__ out, long long total, int f)
{
    long long idx = (long long)blockIdx.x * blockDim.x + threadIdx.x;
    if (idx < total) {
        int j = (int)(idx % f);
        float t = agg[idx] + b[j];
        float y = (t - m[j]) * rsqrtf(v[j] + BN_EPS) * g[j] + be[j];
        out[idx] = y > 0.0f ? y : 0.0f;
    }
}

__global__ void k_add_bias(float* __restrict__ out, const float* __restrict__ b,
                           long long total, int f)
{
    long long idx = (long long)blockIdx.x * blockDim.x + threadIdx.x;
    if (idx < total) out[idx] += b[(int)(idx % f)];
}

// ---------------- host launcher ----------------
extern "C" void kernel_launch(void* const* d_in, const int* in_sizes, int n_in,
                              void* d_out, int out_size, void* d_ws, size_t ws_size,
                              hipStream_t stream)
{
    const float* x   = (const float*)d_in[0];
    const int*   src = (const int*)  d_in[1];
    const int*   dst = (const int*)  d_in[2];
    const float* W1  = (const float*)d_in[3];
    const float* b1  = (const float*)d_in[4];
    const float* g1  = (const float*)d_in[5];
    const float* be1 = (const float*)d_in[6];
    const float* m1  = (const float*)d_in[7];
    const float* v1  = (const float*)d_in[8];
    const float* W2  = (const float*)d_in[9];
    const float* b2  = (const float*)d_in[10];
    const float* g2  = (const float*)d_in[11];
    const float* be2 = (const float*)d_in[12];
    const float* m2  = (const float*)d_in[13];
    const float* v2  = (const float*)d_in[14];
    const float* W3  = (const float*)d_in[15];
    const float* b3  = (const float*)d_in[16];

    const int HID = in_sizes[4];            // 128
    const int C   = in_sizes[16];           // 10
    const int FIN = in_sizes[3] / HID;      // 128
    const int N   = in_sizes[0] / FIN;      // 100000
    const int E   = in_sizes[1];            // 1600000

    const int KS      = FIN / 32;           // 4
    const int tn_hid  = (HID + 15) / 16;    // 8
    const int tn_c    = (C + 15) / 16;      // 1
    const int wf_hid  = KS * tn_hid * 512;  // halves in packed W (hid out)
    const int wf_c    = KS * tn_c * 512;    // halves in packed W (C out)

    // workspace layout (floats): dinv | A | B | H3 | Wf1 | Wf2 | Wf3
    float* ws   = (float*)d_ws;
    size_t off  = 0;
    float* dinv = ws;                 off += ((size_t)N + 255) & ~(size_t)255;
    float* A    = ws + off;           off += (size_t)N * HID;
    float* B    = ws + off;           off += (size_t)N * HID;
    float* H3   = ws + off;           off += ((size_t)N * C + 255) & ~(size_t)255;
    _Float16* Wf1 = (_Float16*)(ws + off);  off += (size_t)wf_hid / 2;
    _Float16* Wf2 = (_Float16*)(ws + off);  off += (size_t)wf_hid / 2;
    _Float16* Wf3 = (_Float16*)(ws + off);
    float* out  = (float*)d_out;

    const int T = 256;
    auto blk = [](long long n, int t) { return (int)((n + t - 1) / t); };

    // ---- pack weights into f16 fragment order ----
    k_pack_w<<<blk(wf_hid, T), T, 0, stream>>>(W1, Wf1, HID, tn_hid, wf_hid);
    k_pack_w<<<blk(wf_hid, T), T, 0, stream>>>(W2, Wf2, HID, tn_hid, wf_hid);
    k_pack_w<<<blk(wf_c,   T), T, 0, stream>>>(W3, Wf3, C,   tn_c,   wf_c);

    // ---- dinv = rsqrt(deg + selfloop) ----
    k_deg_init  <<<blk(N, T), T, 0, stream>>>(dinv, N);
    k_deg_accum <<<blk(E, T), T, 0, stream>>>(dst, dinv, E);
    k_deg_rsqrt <<<blk(N, T), T, 0, stream>>>(dinv, N);

    const long long NH = (long long)N * HID;
    const long long NC = (long long)N * C;
    const int tiles_m  = (N + 15) / 16;
    const int gh = (tiles_m * tn_hid + 7) / 8;   // gemm blocks, hid out
    const int gc = (tiles_m * tn_c + 7) / 8;     // gemm blocks, C out
    const int sc128 = blk((long long)E * 32, T); // scatter blocks, f=128

    // ---- layer 1: x -> A ; agg into B ; relu(bn) in-place ----
    k_gemm_wmma<4><<<gh, T, 0, stream>>>(x, Wf1, A, N, HID, tn_hid);
    k_selfloop_init<<<blk(NH, T), T, 0, stream>>>(A, dinv, B, NH, HID);
    k_edge_scatter128<<<sc128, T, 0, stream>>>(src, dst, dinv, A, B, E);
    k_bias_bn_relu<<<blk(NH, T), T, 0, stream>>>(B, b1, g1, be1, m1, v1, B, NH, HID);

    // ---- layer 2: B -> A ; agg into B ; relu(bn) in-place ----
    k_gemm_wmma<4><<<gh, T, 0, stream>>>(B, Wf2, A, N, HID, tn_hid);
    k_selfloop_init<<<blk(NH, T), T, 0, stream>>>(A, dinv, B, NH, HID);
    k_edge_scatter128<<<sc128, T, 0, stream>>>(src, dst, dinv, A, B, E);
    k_bias_bn_relu<<<blk(NH, T), T, 0, stream>>>(B, b2, g2, be2, m2, v2, B, NH, HID);

    // ---- layer 3: B -> H3 ; agg into d_out ; + b3 ----
    k_gemm_wmma<4><<<gc, T, 0, stream>>>(B, Wf3, H3, N, C, tn_c);
    k_selfloop_init<<<blk(NC, T), T, 0, stream>>>(H3, dinv, out, NC, C);
    k_edge_scatter_small<<<blk(E, T), T, 0, stream>>>(src, dst, dinv, H3, out, E, C);
    k_add_bias<<<blk(NC, T), T, 0, stream>>>(out, b3, NC, C);
}
